// PointSorter_68384469287489
// MI455X (gfx1250) — matrix operand order repaired and checked by reading, not compiled, not run on hardware
//
#include <hip/hip_runtime.h>
#include <hip/hip_bf16.h>
#include <math.h>

typedef __attribute__((ext_vector_type(16))) _Float16 v16h;
typedef __attribute__((ext_vector_type(8)))  _Float16 v8h;
typedef __attribute__((ext_vector_type(8)))  float    v8f;

#define N_PTS   32768
#define C_IN    6
#define H_DIM   64
#define B_SC    4
#define M_SC    8192
#define K_FAR   32
#define K_NEAR  16
#define K_SM    8

// ---- workspace layout (bytes) ----
#define OFF_Z1    0u                          // N*64 f32  (8 MB)
#define OFF_H1H   (OFF_Z1  + N_PTS*H_DIM*4u)  // N*64 f16  (4 MB)
#define OFF_Z2    (OFF_H1H + N_PTS*H_DIM*2u)  // N*64 f32  (8 MB)
#define OFF_S     (OFF_Z2  + N_PTS*H_DIM*4u)  // N f32
#define OFF_IDX   (OFF_S   + N_PTS*4u)        // N*32 i32  (4 MB)
#define OFF_W2T   (OFF_IDX + N_PTS*K_FAR*4u)  // 64*64 f16 (transposed, N-major)
#define OFF_ST1   (OFF_W2T + H_DIM*H_DIM*2u)  // 128 f32 (colsum, colsumsq)
#define OFF_SC1   (OFF_ST1 + 128u*4u)         // 128 f32 (scale, shift)
#define OFF_ST2   (OFF_SC1 + 128u*4u)
#define OFF_SC2   (OFF_ST2 + 128u*4u)
#define OFF_ACC   (OFF_SC2 + 128u*4u)         // 16 f32 accumulators

__device__ __forceinline__ float sigmoidf_(float x) { return 1.0f / (1.0f + expf(-x)); }
__device__ __forceinline__ float geluf_(float x) { return 0.5f * x * (1.0f + erff(x * 0.70710678118654752f)); }

// ---------------- zero the cross-call accumulators ----------------
__global__ void ps_zero(float* acc) {
    if (threadIdx.x < 16) acc[threadIdx.x] = 0.0f;
}

// ---------------- layer 1: z1 = feat @ W1 + b1  (K=6, plain VALU) ----------------
__global__ void ps_gemm1(const float* __restrict__ feat, const float* __restrict__ W1,
                         const float* __restrict__ b1, float* __restrict__ z1) {
    __shared__ float sW[C_IN * H_DIM];
    for (int t = threadIdx.x; t < C_IN * H_DIM; t += blockDim.x) sW[t] = W1[t];
    __syncthreads();
    int gid = blockIdx.x * blockDim.x + threadIdx.x;   // N*64 threads
    int r = gid >> 6, c = gid & 63;
    float acc = b1[c];
    #pragma unroll
    for (int k = 0; k < C_IN; ++k) acc += feat[r * C_IN + k] * sW[k * H_DIM + c];
    z1[gid] = acc;
}

// ---------------- per-column sum / sumsq (one block per column) ----------------
__global__ void ps_colstats(const float* __restrict__ z, float* __restrict__ stats) {
    __shared__ float rs[256], rq[256];
    int col = blockIdx.x, tid = threadIdx.x;
    float s = 0.0f, q = 0.0f;
    for (int r = tid; r < N_PTS; r += 256) {
        float v = z[r * H_DIM + col];
        s += v; q += v * v;
    }
    rs[tid] = s; rq[tid] = q; __syncthreads();
    for (int w = 128; w > 0; w >>= 1) {
        if (tid < w) { rs[tid] += rs[tid + w]; rq[tid] += rq[tid + w]; }
        __syncthreads();
    }
    if (tid == 0) { stats[col] = rs[0]; stats[H_DIM + col] = rq[0]; }
}

// ---------------- fold BN into per-channel scale/shift ----------------
__global__ void ps_bnfin(const float* __restrict__ stats, const float* __restrict__ g,
                         const float* __restrict__ be, float* __restrict__ sc) {
    int c = threadIdx.x;
    if (c < H_DIM) {
        float mean = stats[c] * (1.0f / N_PTS);
        float var  = stats[H_DIM + c] * (1.0f / N_PTS) - mean * mean;
        float scale = g[c] * rsqrtf(var + 1e-5f);
        sc[c] = scale;
        sc[H_DIM + c] = be[c] - mean * scale;
    }
}

// ---------------- W2 f32 -> f16, transposed to N-major for contiguous B fragments ----
__global__ void ps_w2t(const float* __restrict__ W2, _Float16* __restrict__ W2t) {
    int gid = blockIdx.x * blockDim.x + threadIdx.x;
    if (gid < H_DIM * H_DIM) {
        int k = gid >> 6, c = gid & 63;
        W2t[c * H_DIM + k] = (_Float16)W2[gid];
    }
}

// ---------------- BN + GELU -> f16 activations ----------------
__global__ void ps_act1(const float* __restrict__ z1, const float* __restrict__ sc,
                        _Float16* __restrict__ h1h) {
    int gid = blockIdx.x * blockDim.x + threadIdx.x;
    int c = gid & 63;
    float v = z1[gid] * sc[c] + sc[H_DIM + c];
    h1h[gid] = (_Float16)geluf_(v);
}

// ---------------- layer 2: z2 = h1 @ W2 + b2 via WMMA f16 -> f32 ----------------
// One wave per 16x16 output tile; K=64 in two 16x16x32 WMMA steps.
// A row-major, B from transposed W2 -> every fragment is two aligned b128 loads.
__global__ void ps_gemm2_wmma(const _Float16* __restrict__ h1h, const _Float16* __restrict__ W2t,
                              const float* __restrict__ b2, float* __restrict__ z2) {
    int wid  = (blockIdx.x * blockDim.x + threadIdx.x) >> 5;   // global wave id
    int lane = threadIdx.x & 31;
    int tile_m = wid >> 2;          // N/16 row tiles
    int tile_n = wid & 3;           // 64/16 col tiles
    int mrow = tile_m * 16 + (lane & 15);
    int ncol = tile_n * 16 + (lane & 15);
    int kbaseA = (lane >> 4) * 8;   // A fragment: halves {kbase..kbase+7, 16+kbase..}
    int kbaseB = (lane >> 4) * 8;   // B fragment: halves {kbaseB..kbaseB+7, 16+kbaseB..}
    const v8h* Arow = (const v8h*)(h1h + mrow * H_DIM);
    const v8h* Brow = (const v8h*)(W2t + ncol * H_DIM);

    v8f c = {};
    #pragma unroll
    for (int k0 = 0; k0 < H_DIM; k0 += 32) {
        v8h alo = Arow[(k0 + kbaseA) >> 3];
        v8h ahi = Arow[(k0 + 16 + kbaseA) >> 3];
        v8h blo = Brow[(k0 + kbaseB) >> 3];
        v8h bhi = Brow[(k0 + 16 + kbaseB) >> 3];
        v16h a = __builtin_shufflevector(alo, ahi, 0,1,2,3,4,5,6,7,8,9,10,11,12,13,14,15);
        v16h b = __builtin_shufflevector(blo, bhi, 0,1,2,3,4,5,6,7,8,9,10,11,12,13,14,15);
        c = __builtin_amdgcn_wmma_f32_16x16x32_f16(false, a, false, b, (short)0, c, false, false);
    }
    float bias = b2[ncol];
    #pragma unroll
    for (int r = 0; r < 8; ++r) {
        int m = tile_m * 16 + r + (lane >> 4) * 8;
        z2[m * H_DIM + ncol] = c[r] + bias;
    }
}

// ---------------- BN + GELU + layer3 + sigmoid -> s ----------------
__global__ void ps_act2_score(const float* __restrict__ z2, const float* __restrict__ sc,
                              const float* __restrict__ W3, const float* __restrict__ b3,
                              float* __restrict__ s) {
    __shared__ float sW3[H_DIM], sScale[H_DIM], sShift[H_DIM];
    if (threadIdx.x < H_DIM) {
        sW3[threadIdx.x]    = W3[threadIdx.x];
        sScale[threadIdx.x] = sc[threadIdx.x];
        sShift[threadIdx.x] = sc[H_DIM + threadIdx.x];
    }
    __syncthreads();
    int r = blockIdx.x * blockDim.x + threadIdx.x;
    const float4* zp = (const float4*)(z2 + r * H_DIM);
    float acc = b3[0];
    #pragma unroll 4
    for (int c4 = 0; c4 < H_DIM / 4; ++c4) {
        float4 v4 = zp[c4];
        int c = c4 * 4;
        acc += geluf_(v4.x * sScale[c+0] + sShift[c+0]) * sW3[c+0];
        acc += geluf_(v4.y * sScale[c+1] + sShift[c+1]) * sW3[c+1];
        acc += geluf_(v4.z * sScale[c+2] + sShift[c+2]) * sW3[c+2];
        acc += geluf_(v4.w * sScale[c+3] + sShift[c+3]) * sW3[c+3];
    }
    s[r] = sigmoidf_(acc);
}

// ---------------- per-scene KNN: top-32 smallest d2 per query ----------------
// Top-32 lives entirely in VGPRs: statically-indexed arrays + fully unrolled
// branch-free predicated shift-insert (no scratch spill).
__global__ void ps_knn(const float* __restrict__ coords, int* __restrict__ idx) {
    __shared__ float st[384];              // 128 candidates x 3 floats
    int scene = blockIdx.x >> 6;
    int q     = (blockIdx.x & 63) * 128 + threadIdx.x;
    int qg    = scene * M_SC + q;
    float qx = coords[qg * 3 + 0], qy = coords[qg * 3 + 1], qz = coords[qg * 3 + 2];

    float bd[K_FAR]; int bi[K_FAR];
    #pragma unroll
    for (int k = 0; k < K_FAR; ++k) { bd[k] = 3.4e38f; bi[k] = qg; }

    for (int cb = 0; cb < M_SC; cb += 128) {
        const float* tbase = coords + (size_t)(scene * M_SC + cb) * 3;
        for (int t = threadIdx.x; t < 384; t += 128) st[t] = tbase[t];  // coalesced
        __syncthreads();
        if (cb + 128 < M_SC)  // pull next tile toward the caches while we select
            __builtin_prefetch(tbase + 384 + threadIdx.x * 3, 0, 1);
        float dmax = bd[K_FAR - 1];
        for (int cc = 0; cc < 128; ++cc) {
            float dx = qx - st[cc * 3 + 0];
            float dy = qy - st[cc * 3 + 1];
            float dz = qz - st[cc * 3 + 2];
            float d2 = dx * dx + dy * dy + dz * dz;
            if (d2 < dmax) {
                int cand = scene * M_SC + cb + cc;
                // slot k takes bd[k-1] if it shifts, or d2 at the insert point
                #pragma unroll
                for (int k = K_FAR - 1; k >= 1; --k) {
                    float pk = bd[k - 1]; int pi = bi[k - 1];
                    bool up  = pk > d2;
                    float nv = up ? pk : d2;
                    int   ni = up ? pi : cand;
                    bool mod = bd[k] > d2;
                    bd[k] = mod ? nv : bd[k];
                    bi[k] = mod ? ni : bi[k];
                }
                if (bd[0] > d2) { bd[0] = d2; bi[0] = cand; }
                dmax = bd[K_FAR - 1];
            }
        }
        __syncthreads();
    }
    #pragma unroll
    for (int k = 0; k < K_FAR; ++k) idx[qg * K_FAR + k] = bi[k];
}

// ---------------- fused neighbor losses ----------------
__device__ __forceinline__ float ps_block_reduce(float v, float* red) {
    int tid = threadIdx.x;
    red[tid] = v; __syncthreads();
    for (int w = 128; w > 0; w >>= 1) {
        if (tid < w) red[tid] += red[tid + w];
        __syncthreads();
    }
    float r = red[0]; __syncthreads();
    return r;
}

__global__ void ps_loss(const float* __restrict__ s, const float* __restrict__ coords,
                        const int* __restrict__ idx, float* __restrict__ acc) {
    __shared__ float red[256];
    int i = blockIdx.x * blockDim.x + threadIdx.x;
    float si = s[i];
    float cx = coords[i * 3 + 0], cy = coords[i * 3 + 1], cz = coords[i * 3 + 2];
    int nb[K_FAR];
    const int4* ip = (const int4*)(idx + (size_t)i * K_FAR);
    #pragma unroll
    for (int c4 = 0; c4 < K_FAR / 4; ++c4) {          // b128 gather of the index row
        int4 v = ip[c4];
        nb[c4 * 4 + 0] = v.x; nb[c4 * 4 + 1] = v.y;
        nb[c4 * 4 + 2] = v.z; nb[c4 * 4 + 3] = v.w;
    }
    float a_w = 0.0f, a_wsd = 0.0f, a_lp = 0.0f, a_ln = 0.0f, nsum = 0.0f;
    #pragma unroll 4
    for (int k = 0; k < K_NEAR; ++k) {
        int j = nb[k];
        float sj = s[j];
        float sd = fabsf(si - sj);
        float dx = cx - coords[j * 3 + 0], dy = cy - coords[j * 3 + 1], dz = cz - coords[j * 3 + 2];
        float d = sqrtf(fmaxf(dx * dx + dy * dy + dz * dz, 1e-24f));
        float w = expf(-d * 10.0f);              // d / T_LOC, T_LOC=0.1
        a_w   += w;
        a_wsd += w * sd * sd;
        a_lp  += logf(sigmoidf_((1.0f - sd) * 2.0f) + 1e-8f);   // sim/T_CON, T_CON=0.5
        if (k < K_SM) nsum += sj;
    }
    #pragma unroll 4
    for (int k = K_NEAR; k < K_FAR; ++k) {
        float sd = fabsf(si - s[nb[k]]);
        a_ln += logf(1.0f - sigmoidf_((1.0f - sd) * 2.0f) + 1e-8f);
    }
    float dm = si - nsum * (1.0f / K_SM);
    float a_sm = dm * dm;

    float r;
    r = ps_block_reduce(a_w,   red); if (threadIdx.x == 0) atomicAdd(&acc[0], r);
    r = ps_block_reduce(a_wsd, red); if (threadIdx.x == 0) atomicAdd(&acc[1], r);
    r = ps_block_reduce(a_lp,  red); if (threadIdx.x == 0) atomicAdd(&acc[2], r);
    r = ps_block_reduce(a_ln,  red); if (threadIdx.x == 0) atomicAdd(&acc[3], r);
    r = ps_block_reduce(a_sm,  red); if (threadIdx.x == 0) atomicAdd(&acc[4], r);
}

__global__ void ps_finalize(const float* __restrict__ acc, float* __restrict__ out) {
    float loss_loc = acc[1] / fmaxf(acc[0], 1e-8f);
    float inv = 1.0f / (float)(N_PTS * K_NEAR);
    float loss_pos = -acc[2] * inv;
    float loss_neg = -acc[3] * inv;
    float loss_sm  = acc[4] * (1.0f / N_PTS);
    out[0] = loss_loc + 0.5f * (loss_pos + loss_neg) + 0.2f * loss_sm;
}

extern "C" void kernel_launch(void* const* d_in, const int* in_sizes, int n_in,
                              void* d_out, int out_size, void* d_ws, size_t ws_size,
                              hipStream_t stream) {
    const float* feat   = (const float*)d_in[0];
    const float* coords = (const float*)d_in[1];
    const float* W1     = (const float*)d_in[2];
    const float* b1     = (const float*)d_in[3];
    const float* g1     = (const float*)d_in[4];
    const float* be1    = (const float*)d_in[5];
    const float* W2     = (const float*)d_in[6];
    const float* b2     = (const float*)d_in[7];
    const float* g2     = (const float*)d_in[8];
    const float* be2    = (const float*)d_in[9];
    const float* W3     = (const float*)d_in[10];
    const float* b3     = (const float*)d_in[11];

    char* ws = (char*)d_ws;
    float*    z1   = (float*)   (ws + OFF_Z1);
    _Float16* h1h  = (_Float16*)(ws + OFF_H1H);
    float*    z2   = (float*)   (ws + OFF_Z2);
    float*    sbuf = (float*)   (ws + OFF_S);
    int*      idx  = (int*)     (ws + OFF_IDX);
    _Float16* W2t  = (_Float16*)(ws + OFF_W2T);
    float*    st1  = (float*)   (ws + OFF_ST1);
    float*    sc1  = (float*)   (ws + OFF_SC1);
    float*    st2  = (float*)   (ws + OFF_ST2);
    float*    sc2  = (float*)   (ws + OFF_SC2);
    float*    acc  = (float*)   (ws + OFF_ACC);
    float*    out  = (float*)d_out;

    ps_zero<<<1, 64, 0, stream>>>(acc);
    ps_gemm1<<<N_PTS * H_DIM / 256, 256, 0, stream>>>(feat, W1, b1, z1);
    ps_colstats<<<H_DIM, 256, 0, stream>>>(z1, st1);
    ps_bnfin<<<1, 64, 0, stream>>>(st1, g1, be1, sc1);
    ps_w2t<<<(H_DIM * H_DIM + 255) / 256, 256, 0, stream>>>(W2, W2t);
    ps_act1<<<N_PTS * H_DIM / 256, 256, 0, stream>>>(z1, sc1, h1h);
    // (N/16 row tiles) x (64/16 col tiles) waves, 8 waves per 256-thread block
    ps_gemm2_wmma<<<(N_PTS / 16) * (H_DIM / 16) / 8, 256, 0, stream>>>(h1h, W2t, b2, z2);
    ps_colstats<<<H_DIM, 256, 0, stream>>>(z2, st2);
    ps_bnfin<<<1, 64, 0, stream>>>(st2, g2, be2, sc2);
    ps_act2_score<<<N_PTS / 256, 256, 0, stream>>>(z2, sc2, W3, b3, sbuf);
    ps_knn<<<B_SC * (M_SC / 128), 128, 0, stream>>>(coords, idx);
    ps_loss<<<N_PTS / 256, 256, 0, stream>>>(sbuf, coords, idx, acc);
    ps_finalize<<<1, 1, 0, stream>>>(acc, out);
}
